// Attention_68289980007036
// MI455X (gfx1250) — compile-verified, matrix operand order checked by reading, not run on hardware
//
#include <hip/hip_runtime.h>

// ---------------------------------------------------------------------------
// Types for CDNA5 WMMA (wave32, 16x16x32 f16 -> f32)
// ---------------------------------------------------------------------------
typedef _Float16 h2   __attribute__((ext_vector_type(2)));
typedef _Float16 h4v  __attribute__((ext_vector_type(4)));
typedef _Float16 h8   __attribute__((ext_vector_type(8)));
typedef _Float16 v16h __attribute__((ext_vector_type(16)));
typedef float    v8f  __attribute__((ext_vector_type(8)));

#define B_   8
#define N_   1024
#define C_   768
#define H_   12
#define HD_  64
#define SCALE_ 0.125f  // 64^-0.5

// Build a 16-half A/B fragment from two contiguous 16-byte chunks.
// Layout (ISA 7.12.2, 16-bit 16x32): lane group g = lane>>4 holds
// K = g*8 .. g*8+7 (first 8 halves) and K = 16+g*8 .. 23+g*8 (second 8).
__device__ __forceinline__ v16h frag2(const _Float16* p0, const _Float16* p1) {
    h8 lo = *(const h8*)p0;
    h8 hi = *(const h8*)p1;
    return __builtin_shufflevector(lo, hi, 0,1,2,3,4,5,6,7,8,9,10,11,12,13,14,15);
}

__device__ __forceinline__ v8f wmma_f16(v16h a, v16h b, v8f c) {
    // (neg_a, A, neg_b, B, c_mod, C, reuse_a, reuse_b)
    return __builtin_amdgcn_wmma_f32_16x16x32_f16(false, a, false, b, (short)0, c, false, false);
}

// ---------------------------------------------------------------------------
// CDNA5 async global -> LDS copy (tracked on ASYNCcnt), 16 bytes per lane.
// VDST = LDS byte address (low 32 bits of the generic shared pointer: the
// LDS aperture maps addr[31:0] to the LDS offset), VADDR = 64-bit global.
// ---------------------------------------------------------------------------
__device__ __forceinline__ void async_copy_b128(const void* gptr, void* lptr) {
    unsigned      loff = (unsigned)(unsigned long long)(uintptr_t)lptr;
    unsigned long long gaddr = (unsigned long long)(uintptr_t)gptr;
    asm volatile("global_load_async_to_lds_b128 %0, %1, off"
                 :: "v"(loff), "v"(gaddr)
                 : "memory");
}
__device__ __forceinline__ void async_wait0() {
    asm volatile("s_wait_asynccnt 0" ::: "memory");
}

// ---------------------------------------------------------------------------
// Kernel 1: fp32 -> fp16 conversion (vectorized x4)
// ---------------------------------------------------------------------------
__global__ void attn_cvt_f16(const float* __restrict__ src,
                             _Float16* __restrict__ dst, int n) {
    int i = (blockIdx.x * blockDim.x + threadIdx.x) * 4;
    if (i < n) {
        float4 f = *(const float4*)(src + i);
        h4v h;
        h.x = (_Float16)f.x; h.y = (_Float16)f.y;
        h.z = (_Float16)f.z; h.w = (_Float16)f.w;
        *(h4v*)(dst + i) = h;
    }
}

// ---------------------------------------------------------------------------
// Kernel 2: QKV GEMM  [8192 x 768] @ [768 x 2304] -> Q,K,V in [B][H][N][64] f16
// Block tile 128x128, 256 threads = 8 waves, each wave 32x64 (2x4 wmma tiles).
// ---------------------------------------------------------------------------
__global__ __launch_bounds__(256) void attn_gemm_qkv(
        const _Float16* __restrict__ A,   // xh  [8192][768]
        const _Float16* __restrict__ Bm,  // wqkvh [768][2304]
        _Float16* __restrict__ q, _Float16* __restrict__ k, _Float16* __restrict__ v) {
    __shared__ _Float16 Ash[128 * 32];
    __shared__ _Float16 Bsh[128 * 32];   // stored transposed: Bsh[n_local][k_local]

    const int t  = threadIdx.x;
    const int l  = t & 31;
    const int w  = t >> 5;
    const int wm = w & 3;       // 4 row groups of 32
    const int wn = w >> 2;      // 2 col groups of 64
    const int g  = l >> 4;      // lane half-group
    const int ln = l & 15;
    const int rowBase = blockIdx.y * 128;
    const int colBase = blockIdx.x * 128;

    v8f acc[2][4];
    #pragma unroll
    for (int i = 0; i < 2; ++i)
        #pragma unroll
        for (int j = 0; j < 4; ++j)
            #pragma unroll
            for (int r = 0; r < 8; ++r) acc[i][j][r] = 0.0f;

    for (int kk = 0; kk < 768; kk += 32) {
        // Stage A tile 128x32 via async global->LDS (layout-preserving copy)
        {
            int row = t >> 1, kp = (t & 1) * 16;
            const _Float16* s = A + (size_t)(rowBase + row) * 768 + kk + kp;
            _Float16* d = Ash + row * 32 + kp;
            async_copy_b128(s,     d);
            async_copy_b128(s + 8, d + 8);
        }
        // Stage B tile 32x128 transposed into Bsh[n][k]: pair two k-rows so
        // each write is a packed b32 (adjacent k are contiguous in Bsh).
        {
            int k2 = (t >> 4) * 2;          // 0,2,..,30
            int ng = (t & 15) * 8;          // 0,8,..,120
            const _Float16* s0 = Bm + (size_t)(kk + k2) * 2304 + colBase + ng;
            h8 r0 = *(const h8*)s0;
            h8 r1 = *(const h8*)(s0 + 2304);
            #pragma unroll
            for (int j = 0; j < 8; ++j) {
                h2 p; p.x = r0[j]; p.y = r1[j];
                *(h2*)&Bsh[(ng + j) * 32 + k2] = p;
            }
        }
        async_wait0();
        __syncthreads();

        v16h af[2];
        #pragma unroll
        for (int i = 0; i < 2; ++i) {
            int row = 32 * wm + 16 * i + ln;
            af[i] = frag2(&Ash[row * 32 + g * 8], &Ash[row * 32 + 16 + g * 8]);
        }
        #pragma unroll
        for (int j = 0; j < 4; ++j) {
            int col = 64 * wn + 16 * j + ln;
            v16h bf = frag2(&Bsh[col * 32 + g * 8], &Bsh[col * 32 + 16 + g * 8]);
            #pragma unroll
            for (int i = 0; i < 2; ++i) acc[i][j] = wmma_f16(af[i], bf, acc[i][j]);
        }
        __syncthreads();
    }

    // Epilogue: scatter into Q/K/V head-major f16 layout [B][H][N][64].
    // which / h / dBase / bb are lane-independent (tiles never straddle the
    // 768-col q/k/v boundary, the 64-col head boundary, or the 1024-row batch
    // boundary), so compute them from uniform bases -> SALU, no per-lane div.
    const int hl = l >> 4;
    const int bb = rowBase >> 10;              // batch index (block-uniform)
    const int nnBase = rowBase & 1023;
    #pragma unroll
    for (int j = 0; j < 4; ++j) {
        int colU  = colBase + 64 * wn + 16 * j;   // uniform, multiple of 16
        int which = colU / 768;
        int cc    = colU - which * 768;
        int h     = cc >> 6;
        int d     = (cc & 63) + ln;
        _Float16* dst = (which == 0) ? q : ((which == 1) ? k : v);
        _Float16* dhead = dst + (((size_t)(bb * H_ + h)) << 10) * HD_ + d;
        #pragma unroll
        for (int i = 0; i < 2; ++i) {
            #pragma unroll
            for (int r = 0; r < 8; ++r) {
                int nn = nnBase + 32 * wm + 16 * i + r + 8 * hl;
                dhead[(size_t)nn * HD_] = (_Float16)acc[i][j][r];
            }
        }
    }
}

// ---------------------------------------------------------------------------
// Kernel 3: Flash attention. One block = (b, h, 64 query rows), 4 waves,
// each wave owns 16 query rows. Online softmax over 32-key chunks.
// ---------------------------------------------------------------------------
__global__ __launch_bounds__(128) void attn_flash(
        const _Float16* __restrict__ Q, const _Float16* __restrict__ K,
        const _Float16* __restrict__ V, _Float16* __restrict__ O) {
    __shared__ _Float16 Qs[64 * 64];    // [qrow][d]
    __shared__ _Float16 Vt[64 * 32];    // [d][key_local]  (transposed V chunk)
    __shared__ _Float16 Ps[4 * 16 * 32];// per-wave P scratch [16][32]

    const int t  = threadIdx.x;
    const int l  = t & 31;
    const int w  = t >> 5;
    const int g  = l >> 4;
    const int hl = g;
    const int ln = l & 15;
    const int qbase = blockIdx.x * 64;
    const int h = blockIdx.y, b = blockIdx.z;

    const size_t headOff = ((size_t)(b * H_ + h)) * N_ * HD_;
    const _Float16* qh = Q + headOff;
    const _Float16* kh = K + headOff;
    const _Float16* vh = V + headOff;

    // Stage 64x64 Q tile via async global->LDS (straight copy)
    {
        int row = t >> 1, part = (t & 1) * 32;
        const _Float16* s = qh + (size_t)(qbase + row) * HD_ + part;
        _Float16* d = Qs + row * 64 + part;
        #pragma unroll
        for (int j = 0; j < 4; ++j) async_copy_b128(s + 8 * j, d + 8 * j);
    }
    async_wait0();
    __syncthreads();

    v16h aq[2];
    {
        int row = 16 * w + ln;
        aq[0] = frag2(&Qs[row * 64 +  0 + g * 8], &Qs[row * 64 + 16 + g * 8]);
        aq[1] = frag2(&Qs[row * 64 + 32 + g * 8], &Qs[row * 64 + 48 + g * 8]);
    }

    v8f o[4];
    float mrow[8], lsum[8];
    #pragma unroll
    for (int j = 0; j < 4; ++j)
        #pragma unroll
        for (int r = 0; r < 8; ++r) o[j][r] = 0.0f;
    #pragma unroll
    for (int r = 0; r < 8; ++r) { mrow[r] = -1e30f; lsum[r] = 0.0f; }

    for (int kc = 0; kc < N_; kc += 32) {
        __syncthreads();   // previous chunk's Vt reads complete
        // Stage V chunk transposed: Vt[d][key_local]; pair two key-rows so
        // each write is a packed b32.
        {
            int key2 = (t >> 3) * 2;        // 0,2,..,30
            int dg   = (t & 7) * 8;         // 0,8,..,56
            const _Float16* s0 = vh + (size_t)(kc + key2) * HD_ + dg;
            h8 r0 = *(const h8*)s0;
            h8 r1 = *(const h8*)(s0 + HD_);
            #pragma unroll
            for (int j = 0; j < 8; ++j) {
                h2 p; p.x = r0[j]; p.y = r1[j];
                *(h2*)&Vt[(dg + j) * 32 + key2] = p;
            }
        }
        __syncthreads();

        // S = Q K^T for two 16-key tiles (K rows are already B^T layout)
        v8f sv[2];
        #pragma unroll
        for (int kt = 0; kt < 2; ++kt) {
            v8f s = {0.f,0.f,0.f,0.f,0.f,0.f,0.f,0.f};
            const _Float16* kr = kh + (size_t)(kc + 16 * kt + ln) * HD_;
            v16h b0 = frag2(kr +  0 + g * 8, kr + 16 + g * 8);
            s = wmma_f16(aq[0], b0, s);
            v16h b1 = frag2(kr + 32 + g * 8, kr + 48 + g * 8);
            s = wmma_f16(aq[1], b1, s);
            sv[kt] = s * SCALE_;
        }

        // Online softmax (row stats live per 16-lane half-group)
        #pragma unroll
        for (int r = 0; r < 8; ++r) {
            float mx = fmaxf(sv[0][r], sv[1][r]);
            #pragma unroll
            for (int off = 1; off < 16; off <<= 1)
                mx = fmaxf(mx, __shfl_xor(mx, off, 32));
            float newm = fmaxf(mrow[r], mx);
            float corr = __expf(mrow[r] - newm);
            mrow[r] = newm;
            float p0 = __expf(sv[0][r] - newm);
            float p1 = __expf(sv[1][r] - newm);
            float rs = p0 + p1;
            #pragma unroll
            for (int off = 1; off < 16; off <<= 1)
                rs += __shfl_xor(rs, off, 32);
            lsum[r] = lsum[r] * corr + rs;
            o[0][r] *= corr; o[1][r] *= corr; o[2][r] *= corr; o[3][r] *= corr;
            // P (C/D layout) -> LDS [16][32] for A-fragment reshape
            _Float16* pr = Ps + (w * 16 + r + 8 * hl) * 32;
            pr[ln]      = (_Float16)p0;
            pr[16 + ln] = (_Float16)p1;
        }

        // O += P @ V  (per-wave LDS region; wave-ordered DS ops, no barrier)
        {
            const _Float16* pw = Ps + w * 16 * 32;
            v16h ap = frag2(&pw[ln * 32 + g * 8], &pw[ln * 32 + 16 + g * 8]);
            #pragma unroll
            for (int j = 0; j < 4; ++j) {
                const _Float16* vr = Vt + (16 * j + ln) * 32;
                v16h bv = frag2(vr + g * 8, vr + 16 + g * 8);
                o[j] = wmma_f16(ap, bv, o[j]);
            }
        }
    }

    // Normalize and store O as f16 into [B][N][C] at columns h*64 + d
    #pragma unroll
    for (int r = 0; r < 8; ++r) {
        float inv = 1.0f / lsum[r];
        o[0][r] *= inv; o[1][r] *= inv; o[2][r] *= inv; o[3][r] *= inv;
    }
    #pragma unroll
    for (int j = 0; j < 4; ++j) {
        int d = h * HD_ + 16 * j + ln;
        #pragma unroll
        for (int r = 0; r < 8; ++r) {
            int m = qbase + 16 * w + r + 8 * hl;
            O[((size_t)b * N_ + m) * C_ + d] = (_Float16)o[j][r];
        }
    }
}

// ---------------------------------------------------------------------------
// Kernel 4: output projection  [8192 x 768] @ [768 x 768] + bias -> fp32 out
// ---------------------------------------------------------------------------
__global__ __launch_bounds__(256) void attn_gemm_proj(
        const _Float16* __restrict__ A,   // Oh [8192][768]
        const _Float16* __restrict__ Bm,  // wprojh [768][768]
        const float* __restrict__ bias, float* __restrict__ out) {
    __shared__ _Float16 Ash[128 * 32];
    __shared__ _Float16 Bsh[128 * 32];

    const int t  = threadIdx.x;
    const int l  = t & 31;
    const int w  = t >> 5;
    const int wm = w & 3;
    const int wn = w >> 2;
    const int g  = l >> 4;
    const int ln = l & 15;
    const int rowBase = blockIdx.y * 128;
    const int colBase = blockIdx.x * 128;

    v8f acc[2][4];
    #pragma unroll
    for (int i = 0; i < 2; ++i)
        #pragma unroll
        for (int j = 0; j < 4; ++j)
            #pragma unroll
            for (int r = 0; r < 8; ++r) acc[i][j][r] = 0.0f;

    for (int kk = 0; kk < 768; kk += 32) {
        {
            int row = t >> 1, kp = (t & 1) * 16;
            const _Float16* s = A + (size_t)(rowBase + row) * 768 + kk + kp;
            _Float16* d = Ash + row * 32 + kp;
            async_copy_b128(s,     d);
            async_copy_b128(s + 8, d + 8);
        }
        {
            int k2 = (t >> 4) * 2;
            int ng = (t & 15) * 8;
            const _Float16* s0 = Bm + (size_t)(kk + k2) * 768 + colBase + ng;
            h8 r0 = *(const h8*)s0;
            h8 r1 = *(const h8*)(s0 + 768);
            #pragma unroll
            for (int j = 0; j < 8; ++j) {
                h2 p; p.x = r0[j]; p.y = r1[j];
                *(h2*)&Bsh[(ng + j) * 32 + k2] = p;
            }
        }
        async_wait0();
        __syncthreads();

        v16h af[2];
        #pragma unroll
        for (int i = 0; i < 2; ++i) {
            int row = 32 * wm + 16 * i + ln;
            af[i] = frag2(&Ash[row * 32 + g * 8], &Ash[row * 32 + 16 + g * 8]);
        }
        #pragma unroll
        for (int j = 0; j < 4; ++j) {
            int col = 64 * wn + 16 * j + ln;
            v16h bf = frag2(&Bsh[col * 32 + g * 8], &Bsh[col * 32 + 16 + g * 8]);
            #pragma unroll
            for (int i = 0; i < 2; ++i) acc[i][j] = wmma_f16(af[i], bf, acc[i][j]);
        }
        __syncthreads();
    }

    const int hl = l >> 4;
    #pragma unroll
    for (int i = 0; i < 2; ++i) {
        #pragma unroll
        for (int j = 0; j < 4; ++j) {
            int col = colBase + 64 * wn + 16 * j + ln;
            float bv = bias[col];
            #pragma unroll
            for (int r = 0; r < 8; ++r) {
                int m = rowBase + 32 * wm + 16 * i + r + 8 * hl;
                out[(size_t)m * 768 + col] = acc[i][j][r] + bv;
            }
        }
    }
}

// ---------------------------------------------------------------------------
// Host launcher
// ---------------------------------------------------------------------------
extern "C" void kernel_launch(void* const* d_in, const int* in_sizes, int n_in,
                              void* d_out, int out_size, void* d_ws, size_t ws_size,
                              hipStream_t stream) {
    (void)in_sizes; (void)n_in; (void)out_size; (void)ws_size;

    const float* x      = (const float*)d_in[0];  // [8,1024,768]
    const float* w_qkv  = (const float*)d_in[1];  // [768,2304]
    const float* w_proj = (const float*)d_in[2];  // [768,768]
    const float* b_proj = (const float*)d_in[3];  // [768]
    float* out = (float*)d_out;

    // Workspace layout (bytes)
    char* ws = (char*)d_ws;
    const size_t SZ_X    = (size_t)B_ * N_ * C_ * 2;      // 12.6 MB (f16)
    const size_t SZ_WQKV = (size_t)C_ * 3 * C_ * 2;       //  3.5 MB
    const size_t SZ_WPRJ = (size_t)C_ * C_ * 2;           //  1.2 MB
    _Float16* xh     = (_Float16*)(ws);
    _Float16* wqkvh  = (_Float16*)(ws + SZ_X);
    _Float16* wprojh = (_Float16*)(ws + SZ_X + SZ_WQKV);
    _Float16* qh     = (_Float16*)(ws + SZ_X + SZ_WQKV + SZ_WPRJ);
    _Float16* kh     = (_Float16*)((char*)qh + SZ_X);
    _Float16* vh     = (_Float16*)((char*)kh + SZ_X);
    _Float16* oh     = (_Float16*)((char*)vh + SZ_X);     // total ~67.6 MB

    const int nx  = B_ * N_ * C_;        // 6291456
    const int nwq = C_ * 3 * C_;         // 1769472
    const int nwp = C_ * C_;             // 589824

    attn_cvt_f16<<<nx  / (4 * 256), 256, 0, stream>>>(x, xh, nx);
    attn_cvt_f16<<<nwq / (4 * 256), 256, 0, stream>>>(w_qkv, wqkvh, nwq);
    attn_cvt_f16<<<nwp / (4 * 256), 256, 0, stream>>>(w_proj, wprojh, nwp);

    attn_gemm_qkv<<<dim3(3 * C_ / 128, B_ * N_ / 128), 256, 0, stream>>>(
        xh, wqkvh, qh, kh, vh);

    attn_flash<<<dim3(N_ / 64, H_, B_), 128, 0, stream>>>(qh, kh, vh, oh);

    attn_gemm_proj<<<dim3(C_ / 128, B_ * N_ / 128), 256, 0, stream>>>(
        oh, wprojh, b_proj, out);
}
